// ProGenAttention_6674379178483
// MI455X (gfx1250) — matrix operand order repaired and checked by reading, not compiled, not run on hardware
//
#include <hip/hip_runtime.h>
#include <hip/hip_bf16.h>

typedef __attribute__((ext_vector_type(16))) __bf16          v16bf;
typedef __attribute__((ext_vector_type(8)))  float           v8f;
typedef __attribute__((ext_vector_type(8)))  unsigned short  v8u;

#define SEQ   2048
#define HID   4096
#define NQKV  12288
#define NH    16
#define HD    256
#define ROT   64
#define SCALE 0.0625f   // 256^-0.5

// ---- helpers ---------------------------------------------------------------
__device__ __forceinline__ unsigned short f2bf(float f) {
    unsigned int u = __builtin_bit_cast(unsigned int, f);
    u += 0x7FFFu + ((u >> 16) & 1u);            // round-to-nearest-even
    return (unsigned short)(u >> 16);
}

union FragU { struct { v8u lo, hi; } p; v16bf b; };

// two contiguous 16-byte chunks -> one 16-element bf16 fragment
__device__ __forceinline__ v16bf make_frag(const unsigned short* p0,
                                           const unsigned short* p1) {
    FragU f;
    f.p.lo = *(const v8u*)p0;
    f.p.hi = *(const v8u*)p1;
    return f.b;
}

#define WMMA_BF16(A, B, C) \
    __builtin_amdgcn_wmma_f32_16x16x32_bf16(false, (A), false, (B), (short)0, (C), false, false)

// 16B-per-lane async copy, global -> LDS (ASYNCcnt-tracked)
__device__ __forceinline__ void async_cp16(unsigned lds_byte_off,
                                           const unsigned short* gptr) {
    asm volatile("global_load_async_to_lds_b128 %0, %1, off"
                 :: "v"(lds_byte_off), "v"(gptr) : "memory");
}
__device__ __forceinline__ void wait_async0() {
    asm volatile("s_wait_asynccnt 0" ::: "memory");
}

// ---- fp32 -> bf16 bulk convert (8 elems / thread) --------------------------
__global__ __launch_bounds__(256) void cvt_bf16(
        const float* __restrict__ src, unsigned short* __restrict__ dst, int n8) {
    int i = blockIdx.x * blockDim.x + threadIdx.x;
    if (i >= n8) return;
    const float4* s = (const float4*)src + (size_t)i * 2;
    float4 a = s[0], b = s[1];
    union { unsigned short us[8]; uint4 v; } o;
    o.us[0] = f2bf(a.x); o.us[1] = f2bf(a.y); o.us[2] = f2bf(a.z); o.us[3] = f2bf(a.w);
    o.us[4] = f2bf(b.x); o.us[5] = f2bf(b.y); o.us[6] = f2bf(b.z); o.us[7] = f2bf(b.w);
    ((uint4*)dst)[i] = o.v;
}

// ---- GEMM: C[M,N] = A[M,K] * W[N,K]^T, bf16 in, fp32 out -------------------
// Double-buffered LDS, tiles staged with global_load_async_to_lds_b128.
#define BM 128
#define BN 128
#define BK 32
#define LSTR 40   // LDS row stride (ushorts); 80B = multiple of 16B

// issue one 128x32-bf16 tile: 256 threads x 2 iters x 16B
__device__ __forceinline__ void issue_tile(const unsigned short* g0, int K,
                                           unsigned lds_base, int tid) {
    #pragma unroll
    for (int it = 0; it < 2; ++it) {
        int row = (tid >> 2) + it * 64;
        int c   = tid & 3;
        async_cp16(lds_base + (unsigned)(row * LSTR + c * 8) * 2,
                   g0 + row * K + c * 8);
    }
}

__global__ __launch_bounds__(256) void gemm_bf16wmma(
        const unsigned short* __restrict__ A, const unsigned short* __restrict__ W,
        float* __restrict__ C, int M, int N, int K) {
    __shared__ __align__(16) unsigned short sA[2][BM * LSTR];
    __shared__ __align__(16) unsigned short sB[2][BN * LSTR];

    const int tid  = threadIdx.x;
    const int lane = tid & 31, wave = tid >> 5;
    const int mrow = lane & 15, cg = lane >> 4;
    const int n0 = blockIdx.x * BN, m0 = blockIdx.y * BM;
    const int wm = (wave >> 2) * 64;   // 0 or 64
    const int wn = (wave & 3) * 32;    // 0,32,64,96

    // low 32 bits of the flat shared address == LDS byte offset (ISA 10.2)
    const unsigned baseA[2] = { (unsigned)(uintptr_t)&sA[0][0],
                                (unsigned)(uintptr_t)&sA[1][0] };
    const unsigned baseB[2] = { (unsigned)(uintptr_t)&sB[0][0],
                                (unsigned)(uintptr_t)&sB[1][0] };

    const unsigned short* Abase = A + (size_t)m0 * K;
    const unsigned short* Wbase = W + (size_t)n0 * K;

    v8f acc[4][2] = {};

    // prologue: fill buffer 0
    issue_tile(Abase, K, baseA[0], tid);
    issue_tile(Wbase, K, baseB[0], tid);

    int buf = 0;
    for (int k0 = 0; k0 < K; k0 += BK) {
        wait_async0();          // this buffer's tile landed in LDS
        __syncthreads();        // visible to all waves; prev reads done

        if (k0 + BK < K) {      // prefetch next tile into the other buffer
            issue_tile(Abase + k0 + BK, K, baseA[buf ^ 1], tid);
            issue_tile(Wbase + k0 + BK, K, baseB[buf ^ 1], tid);
        }

        const unsigned short* sa = &sA[buf][0];
        const unsigned short* sb = &sB[buf][0];
        v16bf af[4], bf[2];
        #pragma unroll
        for (int mi = 0; mi < 4; ++mi) {
            const unsigned short* r = sa + (wm + mi * 16 + mrow) * LSTR;
            af[mi] = make_frag(r + cg * 8, r + 16 + cg * 8);       // A 16x32 layout
        }
        #pragma unroll
        for (int ni = 0; ni < 2; ++ni) {
            const unsigned short* r = sb + (wn + ni * 16 + mrow) * LSTR;
            bf[ni] = make_frag(r + cg * 8, r + 16 + cg * 8);       // B 32x16 layout
        }
        #pragma unroll
        for (int mi = 0; mi < 4; ++mi)
            #pragma unroll
            for (int ni = 0; ni < 2; ++ni)
                acc[mi][ni] = WMMA_BF16(af[mi], bf[ni], acc[mi][ni]);

        buf ^= 1;
    }

    // 32-bit indexed epilogue (M*N < 2^31 for all our shapes)
    #pragma unroll
    for (int mi = 0; mi < 4; ++mi)
        #pragma unroll
        for (int ni = 0; ni < 2; ++ni)
            #pragma unroll
            for (int r = 0; r < 8; ++r) {
                unsigned idx = (unsigned)(m0 + wm + mi * 16 + r + 8 * cg) * (unsigned)N
                             + (unsigned)(n0 + wn + ni * 16 + mrow);
                C[idx] = acc[mi][ni][r];
            }
}

// ---- RoPE (GPT-J interleaved) + bf16 pack; V stored transposed [h][d][s] ---
__global__ __launch_bounds__(256) void rope_pack(
        const float* __restrict__ qkv, const int* __restrict__ pos,
        unsigned short* __restrict__ Qb, unsigned short* __restrict__ Kb,
        unsigned short* __restrict__ Vt) {
    const int s = blockIdx.x;
    const int d = threadIdx.x;   // 0..255
    const float p = (float)pos[s];
    float c = 1.f, sn = 0.f;
    if (d < ROT) {
        int i = d >> 1;
        // inv_freq = 10000^(-2i/64); ln(10000)=9.210340371976184
        float ang = p * __expf(-(float)(2 * i) * (9.210340371976184f / 64.0f));
        c = __cosf(ang); sn = __sinf(ang);
    }
    const float sgn = (d & 1) ? 1.0f : -1.0f;  // even: x*c - xp*s ; odd: x*c + xp*s
    const float* base = qkv + (size_t)s * NQKV;
    #pragma unroll
    for (int h = 0; h < NH; ++h) {
        const float* qs = base + (h >> 1) * 1536 + (h & 1) * 256;  // q | v | k thirds
        const float* vs = qs + 512;
        const float* ks = qs + 1024;
        float qv = qs[d], kv = ks[d], vv = vs[d];
        if (d < ROT) {
            float qp = qs[d ^ 1], kp = ks[d ^ 1];
            qv = qv * c + sgn * qp * sn;
            kv = kv * c + sgn * kp * sn;
        }
        size_t idx = ((size_t)h * SEQ + s) * HD + d;
        Qb[idx] = f2bf(qv);
        Kb[idx] = f2bf(kv);
        Vt[((size_t)h * HD + d) * SEQ + s] = f2bf(vv);
    }
}

// ---- Flash attention: wave = 16 query rows, 32-key blocks, online softmax --
__global__ __launch_bounds__(128) void flash_attn(
        const unsigned short* __restrict__ Qb, const unsigned short* __restrict__ Kb,
        const unsigned short* __restrict__ Vt, unsigned short* __restrict__ attn) {
    __shared__ __align__(16) unsigned short sP[4][16 * LSTR];   // per-wave P tile

    const int tid  = threadIdx.x;
    const int lane = tid & 31, wave = tid >> 5;
    const int mrow = lane & 15, cg = lane >> 4;
    const int h    = blockIdx.y;
    const int qRow = blockIdx.x * 64 + wave * 16;

    // Q fragments for this wave's 16 rows (full head_dim, 8 chunks of K=32)
    v16bf aq[8];
    const unsigned short* qr = Qb + ((size_t)h * SEQ + qRow + mrow) * HD;
    #pragma unroll
    for (int kc = 0; kc < 8; ++kc)
        aq[kc] = make_frag(qr + kc * 32 + cg * 8, qr + kc * 32 + 16 + cg * 8);

    v8f o[16] = {};
    float mrun[8], lrun[8];
    #pragma unroll
    for (int r = 0; r < 8; ++r) { mrun[r] = -1e30f; lrun[r] = 0.f; }

    unsigned short* pb = sP[wave];
    const int nblk = (qRow + 16 + 31) >> 5;

    for (int blk = 0; blk < nblk; ++blk) {
        const int t0 = blk * 32;
        // S = Q * K^T  (two 16-key column tiles)
        v8f s0 = {}, s1 = {};
        #pragma unroll
        for (int kc = 0; kc < 8; ++kc) {
            const unsigned short* k0 = Kb + ((size_t)h * SEQ + t0 + mrow) * HD + kc * 32 + cg * 8;
            s0 = WMMA_BF16(aq[kc], make_frag(k0, k0 + 16), s0);
            const unsigned short* k1 = Kb + ((size_t)h * SEQ + t0 + 16 + mrow) * HD + kc * 32 + cg * 8;
            s1 = WMMA_BF16(aq[kc], make_frag(k1, k1 + 16), s1);
        }
        // scale + causal mask + online softmax (row r lives in vgpr r, half-wave cg)
        v8f p0, p1;
        #pragma unroll
        for (int r = 0; r < 8; ++r) {
            const int qi = qRow + r + 8 * cg;
            float v0 = s0[r] * SCALE; if (t0 + mrow > qi)      v0 = -3.0e38f;
            float v1 = s1[r] * SCALE; if (t0 + 16 + mrow > qi) v1 = -3.0e38f;
            float rm = fmaxf(v0, v1);
            #pragma unroll
            for (int m = 1; m <= 8; m <<= 1) rm = fmaxf(rm, __shfl_xor(rm, m, 32));
            float mnew  = fmaxf(mrun[r], rm);
            float alpha = __expf(mrun[r] - mnew);
            float e0 = __expf(v0 - mnew), e1 = __expf(v1 - mnew);
            float rs = e0 + e1;
            #pragma unroll
            for (int m = 1; m <= 8; m <<= 1) rs += __shfl_xor(rs, m, 32);
            lrun[r] = lrun[r] * alpha + rs;
            mrun[r] = mnew;
            #pragma unroll
            for (int ni = 0; ni < 16; ++ni) o[ni][r] *= alpha;
            p0[r] = e0; p1[r] = e1;
        }
        // C-layout -> A-layout via per-wave LDS tile (wave DS ops are in-order)
        #pragma unroll
        for (int r = 0; r < 8; ++r) {
            pb[(r + 8 * cg) * LSTR + mrow]      = f2bf(p0[r]);
            pb[(r + 8 * cg) * LSTR + 16 + mrow] = f2bf(p1[r]);
        }
        const unsigned short* pr = pb + mrow * LSTR;
        v16bf ap = make_frag(pr + cg * 8, pr + 16 + cg * 8);
        // O += P * V   (V transposed: contiguous along keys)
        #pragma unroll
        for (int ni = 0; ni < 16; ++ni) {
            const unsigned short* vr =
                Vt + ((size_t)h * HD + ni * 16 + mrow) * SEQ + t0 + cg * 8;
            o[ni] = WMMA_BF16(ap, make_frag(vr, vr + 16), o[ni]);
        }
    }

    // normalize + write attn[s, h*256 + d] as bf16 (feeds bf16 GEMM2 directly)
    #pragma unroll
    for (int r = 0; r < 8; ++r) {
        const float inv = 1.0f / lrun[r];
        const int qi = qRow + r + 8 * cg;
        #pragma unroll
        for (int ni = 0; ni < 16; ++ni)
            attn[(size_t)qi * HID + h * HD + ni * 16 + mrow] = f2bf(o[ni][r] * inv);
    }
}

// ---- host glue -------------------------------------------------------------
extern "C" void kernel_launch(void* const* d_in, const int* in_sizes, int n_in,
                              void* d_out, int out_size, void* d_ws, size_t ws_size,
                              hipStream_t stream) {
    const float* hidden = (const float*)d_in[0];
    const int*   pos    = (const int*)  d_in[1];
    const float* w_qkv  = (const float*)d_in[2];
    const float* w_out  = (const float*)d_in[3];
    float*       out    = (float*)d_out;

    char* ws = (char*)d_ws;
    size_t off = 0;
    auto carve = [&](size_t bytes) -> void* {
        void* p = ws + off;
        off += (bytes + 255) & ~(size_t)255;
        return p;
    };
    unsigned short* hid_bf  = (unsigned short*)carve((size_t)SEQ * HID * 2);
    unsigned short* wqkv_bf = (unsigned short*)carve((size_t)NQKV * HID * 2);  // 100.7 MB
    unsigned short* wout_bf = (unsigned short*)carve((size_t)HID * HID * 2);
    float*          qkv     = (float*)carve((size_t)SEQ * NQKV * 4);           // 100.7 MB
    unsigned short* Qb      = (unsigned short*)carve((size_t)NH * SEQ * HD * 2);
    unsigned short* Kb      = (unsigned short*)carve((size_t)NH * SEQ * HD * 2);
    unsigned short* Vt      = (unsigned short*)carve((size_t)NH * SEQ * HD * 2);
    unsigned short* attn_bf = (unsigned short*)carve((size_t)SEQ * HID * 2);

    const int n8_hid  = SEQ * HID / 8;
    const int n8_wqkv = NQKV * HID / 8;
    const int n8_wout = HID * HID / 8;
    cvt_bf16<<<(n8_hid  + 255) / 256, 256, 0, stream>>>(hidden, hid_bf,  n8_hid);
    cvt_bf16<<<(n8_wqkv + 255) / 256, 256, 0, stream>>>(w_qkv,  wqkv_bf, n8_wqkv);
    cvt_bf16<<<(n8_wout + 255) / 256, 256, 0, stream>>>(w_out,  wout_bf, n8_wout);

    gemm_bf16wmma<<<dim3(NQKV / BN, SEQ / BM), 256, 0, stream>>>(
        hid_bf, wqkv_bf, qkv, SEQ, NQKV, HID);
    rope_pack<<<SEQ, 256, 0, stream>>>(qkv, pos, Qb, Kb, Vt);
    flash_attn<<<dim3(SEQ / 64, NH), 128, 0, stream>>>(Qb, Kb, Vt, attn_bf);

    // final projection: out fp32, bf16 operands
    // reuse GEMM with attn_bf as A
    gemm_bf16wmma<<<dim3(HID / BN, SEQ / BM), 256, 0, stream>>>(
        attn_bf, wout_bf, out, SEQ, HID, HID);
}
